// Attention_90486370992549
// MI455X (gfx1250) — matrix operand order repaired and checked by reading, not compiled, not run on hardware
//
#include <hip/hip_runtime.h>

// Problem constants
#define Bq    128
#define Nq    196
#define Cq    768
#define Hq    12
#define Dq    64
#define NPAD  224                 // 196 padded to 7 chunks of 32 keys
#define BHq   (Bq * Hq)           // 1536
#define SQE   ((long)BHq * NPAD * Dq)   // elems per Q/K/Vt buffer = 22020096
#define ROWS  (Bq * Nq)           // 25088 (divisible by 128)
#define QKSCALE 0.125f            // 64^-0.5

typedef __attribute__((ext_vector_type(16))) _Float16 v16h;
typedef __attribute__((ext_vector_type(8)))  float    v8f;
typedef __attribute__((ext_vector_type(8)))  _Float16 h8;
typedef __attribute__((ext_vector_type(2)))  _Float16 h2;
typedef __attribute__((ext_vector_type(4)))  unsigned u32x4;
typedef __attribute__((ext_vector_type(8)))  int      i32x8;
typedef __attribute__((ext_vector_type(4)))  int      i32x4;

// ---------------------------------------------------------------------------
// WMMA helper (CDNA5 16x16x32 f16 -> f32)
// ---------------------------------------------------------------------------
static __device__ __forceinline__ v8f wmma_f16(v16h a, v16h b, v8f c) {
  return __builtin_amdgcn_wmma_f32_16x16x32_f16(
      /*neg_a=*/false, a, /*neg_b=*/false, b,
      /*c_mod=*/(short)0, c, /*reuse_a=*/false, /*reuse_b=*/false);
}

// ---------------------------------------------------------------------------
// Tensor Data Mover: 2D f16 tile (tile_w x tile_h elems, row stride
// stride_elems) from global -> LDS.  D# built per CDNA5 ISA 8.3/8.4.
// ---------------------------------------------------------------------------
static __device__ __forceinline__ void tdm_load_2d_f16(
    unsigned lds_off, const void* gptr,
    unsigned tile_w, unsigned tile_h, unsigned stride_elems) {
  const unsigned long long ga = (unsigned long long)(uintptr_t)gptr;
  u32x4 g0 = {};
  g0[0] = 1u;                                           // count=1, user mode
  g0[1] = lds_off;                                      // lds_addr
  g0[2] = (unsigned)ga;                                 // global_addr[31:0]
  g0[3] = (unsigned)((ga >> 32) & 0x01ffffffu) | (2u << 30); // addr hi | type=2
  i32x8 g1 = {};
  g1[0] = (1 << 16);                                    // data_size=2B, mask=0
  g1[1] = (int)((tile_w & 0xffffu) << 16);              // tensor_dim0[15:0]
  g1[2] = (int)((tile_w >> 16) | ((tile_h & 0xffffu) << 16)); // d0 hi | d1 lo
  g1[3] = (int)((tile_h >> 16) | (tile_w << 16));       // d1 hi | tile_dim0
  g1[4] = (int)(tile_h & 0xffffu);                      // tile_dim1 (dim2=0)
  g1[5] = (int)stride_elems;                            // dim0_stride[31:0]
  g1[6] = 0;                                            // stride hi bits
  g1[7] = 0;
  const i32x4 z4 = {};
#if defined(__clang_major__) && (__clang_major__ >= 23)
  const i32x8 z8 = {};
  __builtin_amdgcn_tensor_load_to_lds(g0, g1, z4, z4, z8, 0);
#else
  __builtin_amdgcn_tensor_load_to_lds(g0, g1, z4, z4, 0);
#endif
}

static __device__ __forceinline__ unsigned lds_offset_of(const void* p) {
  return (unsigned)(unsigned long long)(uintptr_t)p;    // aperture low 32 bits
}

// A-matrix fragment (16x32 f16, row-major source, leading dim `ld`).
// Lane L -> M = L%16; VGPR i holds K pair kk = 16*(i/4) + 2*(i%4) + 8*(L/16).
static __device__ __forceinline__ v16h load_frag_a(const _Float16* src, int ld, int lane) {
  const int m  = lane & 15;
  const int kh = (lane >> 4) & 1;
  const _Float16* p = src + m * ld;
  v16h a;
#pragma unroll
  for (int i = 0; i < 8; ++i) {
    const int kk = ((i >> 2) << 4) + ((i & 3) << 1) + (kh << 3);
    h2 t = *(const h2*)(p + kk);
    a[2 * i]     = t[0];
    a[2 * i + 1] = t[1];
  }
  return a;
}

// B-matrix fragment (32x16) where B[k][n] = src[n][k]  (src row-major, ld).
// Lane L -> N = L%16; elems 0-7 K=8*(L/16)+0..7, elems 8-15 K=16+8*(L/16)+0..7.
static __device__ __forceinline__ v16h load_frag_bT(const _Float16* src, int ld, int lane) {
  const int n  = lane & 15;
  const int kh = (lane >> 4) & 1;
  const h8* p = (const h8*)(src + n * ld + (kh << 3));
  h8 lo = p[0];
  h8 hi = p[2];
  v16h b;
#pragma unroll
  for (int i = 0; i < 8; ++i) { b[i] = lo[i]; b[8 + i] = hi[i]; }
  return b;
}

static __device__ __forceinline__ float redmax16(float v) {
  v = fmaxf(v, __shfl_xor(v, 1, 32));
  v = fmaxf(v, __shfl_xor(v, 2, 32));
  v = fmaxf(v, __shfl_xor(v, 4, 32));
  v = fmaxf(v, __shfl_xor(v, 8, 32));
  return v;
}
static __device__ __forceinline__ float redsum16(float v) {
  v += __shfl_xor(v, 1, 32);
  v += __shfl_xor(v, 2, 32);
  v += __shfl_xor(v, 4, 32);
  v += __shfl_xor(v, 8, 32);
  return v;
}

// ---------------------------------------------------------------------------
// Kernel 0: zero-fill workspace (padded Q/K/Vt regions must read as 0)
// ---------------------------------------------------------------------------
__global__ void zero_ws_kernel(uint4* __restrict__ p, long n4) {
  long i = (long)blockIdx.x * blockDim.x + threadIdx.x;
  if (i < n4) p[i] = make_uint4(0u, 0u, 0u, 0u);
}

// ---------------------------------------------------------------------------
// Kernel 1: QKV GEMM.  qkv = x @ Wqkv^T + b  -> scatter into Q,K (row-major)
// and Vt (transposed).  Block 256 = 8 waves; 128x64 tile; each wave owns a
// 32x32 sub-tile -> 4 WMMAs per 32-K chunk from 2 A + 2 B fragments.
// ---------------------------------------------------------------------------
__global__ __launch_bounds__(256) void qkv_gemm_kernel(
    const float* __restrict__ x, const float* __restrict__ Wqkv,
    const float* __restrict__ bqkv,
    _Float16* __restrict__ Qw, _Float16* __restrict__ Kw,
    _Float16* __restrict__ Vt) {
  __shared__ _Float16 xs[128][32];
  __shared__ _Float16 wsm[64][32];

  const int row0 = blockIdx.x * 128;
  const int o0   = blockIdx.y * 64;
  const int tid  = threadIdx.x;
  const int lane = tid & 31;
  const int wave = tid >> 5;
  const int wr   = wave >> 1;        // 0..3 : 32-row strip
  const int wc   = wave & 1;         // 0..1 : 32-col strip

  const int xrow = tid >> 1;         // 0..127
  const int xcb  = (tid & 1) << 4;   // 0,16
  const int wrow = tid >> 2;         // 0..63
  const int wcb  = (tid & 3) << 3;   // 0,8,16,24

  v8f acc[2][2] = {{v8f{}, v8f{}}, {v8f{}, v8f{}}};

  for (int kc = 0; kc < Cq; kc += 32) {
    const float4* xp = (const float4*)(x    + (long)(row0 + xrow) * Cq + kc + xcb);
    const float4* wp = (const float4*)(Wqkv + (long)(o0   + wrow) * Cq + kc + wcb);
    float4 x0 = xp[0], x1 = xp[1], x2 = xp[2], x3 = xp[3];
    float4 w0 = wp[0], w1 = wp[1];
    if (kc + 32 < Cq) {
      __builtin_prefetch(xp + 8, 0, 1);
      __builtin_prefetch(wp + 8, 0, 1);
    }
    __syncthreads();
    {
      h8 v;
      v[0]=(_Float16)x0.x; v[1]=(_Float16)x0.y; v[2]=(_Float16)x0.z; v[3]=(_Float16)x0.w;
      v[4]=(_Float16)x1.x; v[5]=(_Float16)x1.y; v[6]=(_Float16)x1.z; v[7]=(_Float16)x1.w;
      *(h8*)&xs[xrow][xcb] = v;
      v[0]=(_Float16)x2.x; v[1]=(_Float16)x2.y; v[2]=(_Float16)x2.z; v[3]=(_Float16)x2.w;
      v[4]=(_Float16)x3.x; v[5]=(_Float16)x3.y; v[6]=(_Float16)x3.z; v[7]=(_Float16)x3.w;
      *(h8*)&xs[xrow][xcb + 8] = v;
      v[0]=(_Float16)w0.x; v[1]=(_Float16)w0.y; v[2]=(_Float16)w0.z; v[3]=(_Float16)w0.w;
      v[4]=(_Float16)w1.x; v[5]=(_Float16)w1.y; v[6]=(_Float16)w1.z; v[7]=(_Float16)w1.w;
      *(h8*)&wsm[wrow][wcb] = v;
    }
    __syncthreads();

    v16h a0 = load_frag_a(&xs[wr * 32][0],      32, lane);
    v16h a1 = load_frag_a(&xs[wr * 32 + 16][0], 32, lane);
    v16h b0 = load_frag_bT(&wsm[wc * 32][0],      32, lane);
    v16h b1 = load_frag_bT(&wsm[wc * 32 + 16][0], 32, lane);
    acc[0][0] = wmma_f16(a0, b0, acc[0][0]);
    acc[0][1] = wmma_f16(a0, b1, acc[0][1]);
    acc[1][0] = wmma_f16(a1, b0, acc[1][0]);
    acc[1][1] = wmma_f16(a1, b1, acc[1][1]);
  }

  // Epilogue: C layout lane L -> N = L%16, VGPR r -> M = r + 8*(L/16)
  const int ncol = lane & 15;
  const int kh   = lane >> 4;
#pragma unroll
  for (int tC = 0; tC < 2; ++tC) {
    const int o = o0 + wc * 32 + tC * 16 + ncol;     // 0..2303
    const float bias = bqkv[o];
    const int qkv_idx = o / Cq;                      // 0=q 1=k 2=v
    const int rem = o - qkv_idx * Cq;
    const int h   = rem >> 6;
    const int d   = rem & 63;
#pragma unroll
    for (int tR = 0; tR < 2; ++tR) {
#pragma unroll
      for (int r = 0; r < 8; ++r) {
        const int g = row0 + wr * 32 + tR * 16 + r + (kh << 3);
        const int b = g / Nq;
        const int n = g - b * Nq;
        const int bh = b * Hq + h;
        const _Float16 hv = (_Float16)(acc[tR][tC][r] + bias);
        if (qkv_idx == 0)      Qw[(long)bh * NPAD * Dq + n * Dq + d] = hv;
        else if (qkv_idx == 1) Kw[(long)bh * NPAD * Dq + n * Dq + d] = hv;
        else                   Vt[(long)bh * Dq * NPAD + d * NPAD + n] = hv;
      }
    }
  }
}

// ---------------------------------------------------------------------------
// Kernel 2: flash attention, one wave per (b,h, 16-query-row tile).
// K and V^T chunks are staged into LDS by the Tensor Data Mover.
// ---------------------------------------------------------------------------
__global__ __launch_bounds__(32) void attn_kernel(
    const _Float16* __restrict__ Qw, const _Float16* __restrict__ Kw,
    const _Float16* __restrict__ Vt, const float* __restrict__ amask,
    _Float16* __restrict__ AO) {
  __shared__ _Float16 Ks[32][64];   // keys kb..kb+31, row-major [key][d]
  __shared__ _Float16 Vs[64][32];   // V^T slice      [d][key-kb]
  __shared__ _Float16 pT[16][32];   // P re-layout (C-layout -> A-layout)

  const int bid  = blockIdx.x;
  const int qt   = bid % 13;
  const int bh   = bid / 13;
  const int b    = bh / Hq;
  const int h    = bh - b * Hq;
  const int lane = threadIdx.x;
  const int q0   = qt * 16;
  const int nlo  = lane & 15;
  const int kh   = lane >> 4;

  const _Float16* Qb = Qw + (long)bh * NPAD * Dq;
  const _Float16* Kb = Kw + (long)bh * NPAD * Dq;
  const _Float16* Vb = Vt + (long)bh * Dq * NPAD;
  const float*    mb = amask + (long)h * Nq * Nq;

  const v16h qa0 = load_frag_a(Qb + q0 * Dq,      Dq, lane);  // K-dim 0..31
  const v16h qa1 = load_frag_a(Qb + q0 * Dq + 32, Dq, lane);  // K-dim 32..63

  v8f O0 = {}, O1 = {}, O2 = {}, O3 = {};
  float mrow[8], lrow[8];
#pragma unroll
  for (int r = 0; r < 8; ++r) { mrow[r] = -1e30f; lrow[r] = 0.f; }

  const unsigned ksOff = lds_offset_of(&Ks[0][0]);
  const unsigned vsOff = lds_offset_of(&Vs[0][0]);

  for (int kb = 0; kb < NPAD; kb += 32) {
    // TDM: stage K rows [kb,kb+32) x 64 and Vt rows [0,64) x cols [kb,kb+32)
    tdm_load_2d_f16(ksOff, Kb + (long)kb * Dq, /*w=*/Dq, /*h=*/32, /*stride=*/Dq);
    tdm_load_2d_f16(vsOff, Vb + kb,            /*w=*/32, /*h=*/Dq, /*stride=*/NPAD);
    __builtin_amdgcn_s_wait_tensorcnt(0);

    v8f S0 = {}, S1 = {};
    {
      v16h bk;
      bk = load_frag_bT(&Ks[0][0],       Dq, lane); S0 = wmma_f16(qa0, bk, S0);
      bk = load_frag_bT(&Ks[0][0] + 32,  Dq, lane); S0 = wmma_f16(qa1, bk, S0);
      bk = load_frag_bT(&Ks[16][0],      Dq, lane); S1 = wmma_f16(qa0, bk, S1);
      bk = load_frag_bT(&Ks[16][0] + 32, Dq, lane); S1 = wmma_f16(qa1, bk, S1);
    }

    const int c0 = kb + nlo;
    const int c1 = kb + 16 + nlo;
#pragma unroll
    for (int r = 0; r < 8; ++r) {
      const int qrow = q0 + r + (kh << 3);
      float s0 = -1e30f, s1 = -1e30f;
      if (qrow < Nq && c0 < Nq) {
        const float mk = mb[qrow * Nq + c0];
        s0 = S0[r] * QKSCALE * (1.f / (1.f + __expf(-mk)));
      }
      if (qrow < Nq && c1 < Nq) {
        const float mk = mb[qrow * Nq + c1];
        s1 = S1[r] * QKSCALE * (1.f / (1.f + __expf(-mk)));
      }
      const float rm   = redmax16(fmaxf(s0, s1));
      const float newm = fmaxf(mrow[r], rm);
      const float corr = __expf(mrow[r] - newm);
      const float p0   = __expf(s0 - newm);
      const float p1   = __expf(s1 - newm);
      const float rsum = redsum16(p0 + p1);
      lrow[r] = lrow[r] * corr + rsum;
      mrow[r] = newm;
      O0[r] *= corr; O1[r] *= corr; O2[r] *= corr; O3[r] *= corr;
      const int m = r + (kh << 3);
      pT[m][nlo]      = (_Float16)p0;
      pT[m][16 + nlo] = (_Float16)p1;
    }
    __syncthreads();
    const v16h pa = load_frag_a(&pT[0][0], 32, lane);
    O0 = wmma_f16(pa, load_frag_bT(&Vs[ 0][0], 32, lane), O0);
    O1 = wmma_f16(pa, load_frag_bT(&Vs[16][0], 32, lane), O1);
    O2 = wmma_f16(pa, load_frag_bT(&Vs[32][0], 32, lane), O2);
    O3 = wmma_f16(pa, load_frag_bT(&Vs[48][0], 32, lane), O3);
    __syncthreads();
  }

  // Epilogue: AO[b, n, h*64 + d]  (f16, feeds projection GEMM row-major)
#pragma unroll
  for (int r = 0; r < 8; ++r) {
    const int n = q0 + r + (kh << 3);
    if (n < Nq) {
      const float inv = 1.f / lrow[r];
      const long base = ((long)(b * Nq + n)) * Cq + h * Dq + nlo;
      AO[base +  0] = (_Float16)(O0[r] * inv);
      AO[base + 16] = (_Float16)(O1[r] * inv);
      AO[base + 32] = (_Float16)(O2[r] * inv);
      AO[base + 48] = (_Float16)(O3[r] * inv);
    }
  }
}

// ---------------------------------------------------------------------------
// Kernel 3: projection GEMM.  out = AO @ Wproj^T + b  (f32 out).
// A-tile (f16) staged by TDM from wave 0; W converted f32->f16 by all threads.
// ---------------------------------------------------------------------------
__global__ __launch_bounds__(256) void proj_gemm_kernel(
    const _Float16* __restrict__ AO, const float* __restrict__ Wp,
    const float* __restrict__ bp, float* __restrict__ out) {
  __shared__ _Float16 xs[128][32];
  __shared__ _Float16 wsm[64][32];

  const int row0 = blockIdx.x * 128;
  const int o0   = blockIdx.y * 64;
  const int tid  = threadIdx.x;
  const int lane = tid & 31;
  const int wave = tid >> 5;
  const int wr   = wave >> 1;
  const int wc   = wave & 1;
  const int wrow = tid >> 2;
  const int wcb  = (tid & 3) << 3;

  const unsigned xsOff = lds_offset_of(&xs[0][0]);

  v8f acc[2][2] = {{v8f{}, v8f{}}, {v8f{}, v8f{}}};

  for (int kc = 0; kc < Cq; kc += 32) {
    __syncthreads();  // previous chunk's fragments fully consumed
    if (wave == 0)    // TDM: 128x32 f16 A-tile straight into LDS
      tdm_load_2d_f16(xsOff, AO + (long)row0 * Cq + kc, 32, 128, Cq);

    const float4* wp = (const float4*)(Wp + (long)(o0 + wrow) * Cq + kc + wcb);
    float4 w0 = wp[0], w1 = wp[1];
    if (kc + 32 < Cq) __builtin_prefetch(wp + 8, 0, 1);
    {
      h8 v;
      v[0]=(_Float16)w0.x; v[1]=(_Float16)w0.y; v[2]=(_Float16)w0.z; v[3]=(_Float16)w0.w;
      v[4]=(_Float16)w1.x; v[5]=(_Float16)w1.y; v[6]=(_Float16)w1.z; v[7]=(_Float16)w1.w;
      *(h8*)&wsm[wrow][wcb] = v;
    }
    if (wave == 0) __builtin_amdgcn_s_wait_tensorcnt(0);
    __syncthreads();

    v16h a0 = load_frag_a(&xs[wr * 32][0],      32, lane);
    v16h a1 = load_frag_a(&xs[wr * 32 + 16][0], 32, lane);
    v16h b0 = load_frag_bT(&wsm[wc * 32][0],      32, lane);
    v16h b1 = load_frag_bT(&wsm[wc * 32 + 16][0], 32, lane);
    acc[0][0] = wmma_f16(a0, b0, acc[0][0]);
    acc[0][1] = wmma_f16(a0, b1, acc[0][1]);
    acc[1][0] = wmma_f16(a1, b0, acc[1][0]);
    acc[1][1] = wmma_f16(a1, b1, acc[1][1]);
  }

  const int ncol = lane & 15;
  const int kh   = lane >> 4;
#pragma unroll
  for (int tC = 0; tC < 2; ++tC) {
    const int o = o0 + wc * 32 + tC * 16 + ncol;
    const float bias = bp[o];
#pragma unroll
    for (int tR = 0; tR < 2; ++tR) {
#pragma unroll
      for (int r = 0; r < 8; ++r) {
        const int g = row0 + wr * 32 + tR * 16 + r + (kh << 3);
        out[(long)g * Cq + o] = acc[tR][tC][r] + bias;
      }
    }
  }
}

// ---------------------------------------------------------------------------
// Launcher
// ---------------------------------------------------------------------------
extern "C" void kernel_launch(void* const* d_in, const int* in_sizes, int n_in,
                              void* d_out, int out_size, void* d_ws, size_t ws_size,
                              hipStream_t stream) {
  (void)in_sizes; (void)n_in; (void)out_size; (void)ws_size;
  const float* x      = (const float*)d_in[0];
  const float* W_qkv  = (const float*)d_in[1];
  const float* b_qkv  = (const float*)d_in[2];
  const float* amask  = (const float*)d_in[3];
  const float* W_proj = (const float*)d_in[4];
  const float* b_proj = (const float*)d_in[5];
  float* out = (float*)d_out;

  _Float16* Qw = (_Float16*)d_ws;
  _Float16* Kw = Qw + SQE;
  _Float16* Vt = Kw + SQE;
  _Float16* AO = Vt + SQE;

  const long totalHalf = 3 * SQE + (long)ROWS * Cq;     // 85,327,872 f16
  const long n4 = (totalHalf * 2 + 15) / 16;            // uint4 count
  zero_ws_kernel<<<(int)((n4 + 255) / 256), 256, 0, stream>>>((uint4*)d_ws, n4);

  qkv_gemm_kernel<<<dim3(ROWS / 128, (3 * Cq) / 64), 256, 0, stream>>>(
      x, W_qkv, b_qkv, Qw, Kw, Vt);

  attn_kernel<<<BHq * 13, 32, 0, stream>>>(Qw, Kw, Vt, amask, AO);

  proj_gemm_kernel<<<dim3(ROWS / 128, Cq / 64), 256, 0, stream>>>(
      AO, W_proj, b_proj, out);
}
